// Attention_56083682951967
// MI455X (gfx1250) — compile-verified, matrix operand order checked by reading, not run on hardware
//
#include <hip/hip_runtime.h>

typedef __attribute__((ext_vector_type(16))) _Float16 v16h;
typedef __attribute__((ext_vector_type(8)))  _Float16 v8h;
typedef __attribute__((ext_vector_type(8)))  float    v8f;
typedef __attribute__((ext_vector_type(4)))  float    v4f;

union Frag16 { v16h v; v8h h[2]; };

__device__ __forceinline__ v8f wmma_f16(v16h a, v16h b, v8f c) {
  return __builtin_amdgcn_wmma_f32_16x16x32_f16(false, a, false, b, (short)0, c,
                                                false, false);
}

// CDNA5 async global->LDS copy (ASYNCcnt-tracked, no VGPR bounce).
__device__ __forceinline__ void async_load_b128(unsigned lds_off, const void* gaddr) {
  asm volatile("global_load_async_to_lds_b128 %0, %1, off"
               :: "v"(lds_off), "v"(gaddr)
               : "memory");
}
__device__ __forceinline__ void wait_async0() {
  asm volatile("s_wait_asynccnt 0x0" ::: "memory");
}

// ---------------------------------------------------------------------------
// Tiled GEMM: C[MxN] = A[MxK] * B[KxN], double-buffered LDS.
//   A: fp32 or f16 (row-major), B: fp32 (row-major), C: f16 or fp32.
//   Block = 256 threads (8 waves), tile 128x64, K-step 32.
//   Wave owns a 32x32 sub-tile (2x2 WMMA grid) -> 4 wmma per K-step.
//   One barrier per K-step; next tile's loads overlap current tile's WMMAs.
// ---------------------------------------------------------------------------
template <bool A_HALF, bool OUT_FLOAT>
__global__ __launch_bounds__(256) void gemm_tile(const void* __restrict__ Av,
                                                 const float* __restrict__ B,
                                                 void* __restrict__ Cv,
                                                 int M, int N, int K) {
  __shared__ _Float16 Al[2][128][32];   // [buf][m][k]  (8 KB per buffer)
  __shared__ _Float16 Bl[2][64][32];    // [buf][n][k]  (4 KB per buffer)

  const int t    = threadIdx.x;
  const int lane = t & 31;
  const int w    = t >> 5;
  const int mt   = w >> 1;              // 0..3 -> 32-row strip
  const int nt   = w & 1;               // 0..1 -> 32-col strip
  const int m0   = blockIdx.y * 128;
  const int n0   = blockIdx.x * 64;
  const int ln   = lane & 15;
  const int off  = (lane & 16) ? 8 : 0; // lo/hi K-half per ISA A/B layout

  const unsigned albase = (unsigned)(uintptr_t)&Al[0][0][0];  // buf stride 8192 B

  auto stageA = [&](int k0, int p) {
    if (A_HALF) {
      const _Float16* A = (const _Float16*)Av;
#pragma unroll
      for (int i = 0; i < 2; ++i) {
        int c = t + 256 * i;
        int row = c >> 2, c8 = c & 3;
        async_load_b128(albase + (unsigned)p * 8192u + row * 64 + c8 * 16,
                        &A[(size_t)(m0 + row) * K + k0 + c8 * 8]);
      }
    } else {
      const float* A = (const float*)Av;
#pragma unroll
      for (int i = 0; i < 4; ++i) {
        int c = t + 256 * i;
        int row = c >> 3, c4 = c & 7;
        v4f a4 = *(const v4f*)&A[(size_t)(m0 + row) * K + k0 + c4 * 4];
#pragma unroll
        for (int j = 0; j < 4; ++j) Al[p][row][c4 * 4 + j] = (_Float16)a4[j];
      }
    }
  };
  auto stageB = [&](int k0, int p) {
#pragma unroll
    for (int i = 0; i < 2; ++i) {
      int c = t + 256 * i;
      int kk = c >> 4, c4 = c & 15;
      v4f b4 = *(const v4f*)&B[(size_t)(k0 + kk) * N + n0 + c4 * 4];
#pragma unroll
      for (int j = 0; j < 4; ++j) Bl[p][c4 * 4 + j][kk] = (_Float16)b4[j];
    }
  };

  v8f acc[2][2] = {};
  stageA(0, 0);
  stageB(0, 0);
  int p = 0;

  for (int k0 = 0; k0 < K; k0 += 32) {
    if (A_HALF) wait_async0();
    __syncthreads();                       // tile p ready; buffer p^1 free
    if (k0 + 32 < K) {                     // prefetch next tile into p^1
      stageA(k0 + 32, p ^ 1);
      stageB(k0 + 32, p ^ 1);
    }
    Frag16 fa[2], fb[2];
#pragma unroll
    for (int i = 0; i < 2; ++i) {
      fa[i].h[0] = *(const v8h*)&Al[p][mt * 32 + i * 16 + ln][off];
      fa[i].h[1] = *(const v8h*)&Al[p][mt * 32 + i * 16 + ln][16 + off];
      fb[i].h[0] = *(const v8h*)&Bl[p][nt * 32 + i * 16 + ln][off];
      fb[i].h[1] = *(const v8h*)&Bl[p][nt * 32 + i * 16 + ln][16 + off];
    }
#pragma unroll
    for (int i = 0; i < 2; ++i)
#pragma unroll
      for (int j = 0; j < 2; ++j)
        acc[i][j] = wmma_f16(fa[i].v, fb[j].v, acc[i][j]);
    p ^= 1;
  }

#pragma unroll
  for (int i = 0; i < 2; ++i)
#pragma unroll
    for (int j = 0; j < 2; ++j)
#pragma unroll
      for (int v = 0; v < 8; ++v) {
        int row = m0 + mt * 32 + i * 16 + v + ((lane & 16) ? 8 : 0);
        int col = n0 + nt * 32 + j * 16 + ln;
        if (OUT_FLOAT)
          ((float*)Cv)[(size_t)row * N + col] = acc[i][j][v];
        else
          ((_Float16*)Cv)[(size_t)row * N + col] = (_Float16)acc[i][j][v];
      }
}

// ---------------------------------------------------------------------------
// RoPE (interleaved pairs), f16 storage, f32 math. scale folds 1/sqrt(dh).
// ---------------------------------------------------------------------------
__global__ __launch_bounds__(256) void rope_kernel(_Float16* __restrict__ x,
                                                   const float* __restrict__ freqs,
                                                   int n_heads, float scale) {
  const int bs = blockIdx.x;
  const int npairs = n_heads * 32;
  for (int idx = threadIdx.x; idx < npairs; idx += 256) {
    int hh = idx >> 5, i = idx & 31;
    size_t base = ((size_t)bs * n_heads + hh) * 64;
    float fc = freqs[(size_t)bs * 64 + i];
    float fs = freqs[(size_t)bs * 64 + 32 + i];
    float xr = (float)x[base + 2 * i];
    float xi = (float)x[base + 2 * i + 1];
    x[base + 2 * i]     = (_Float16)((xr * fc - xi * fs) * scale);
    x[base + 2 * i + 1] = (_Float16)((xr * fs + xi * fc) * scale);
  }
}

// ---------------------------------------------------------------------------
// Flash attention, one (b, head) per blockIdx.y, 64 query rows per block.
//   128 threads = 4 waves; wave w owns query rows [w*16, w*16+16).
//   KV-block = 64 columns: softmax butterflies, acc rescale, barriers and
//   async waits amortize over 2x the columns; WMMA work per column unchanged.
//   Double-buffered tiles, ONE barrier per KV-block; K + mask async to LDS.
// ---------------------------------------------------------------------------
__global__ __launch_bounds__(128) void attn_kernel(const _Float16* __restrict__ Q,
                                                   const _Float16* __restrict__ Kx,
                                                   const _Float16* __restrict__ Vx,
                                                   const float* __restrict__ mask,
                                                   _Float16* __restrict__ Out) {
  __shared__ _Float16 Kl[2][64][64];     // [buf][kv][d]  — B layout for QK^T (16 KB)
  __shared__ _Float16 Vt[2][64][64];     // [buf][d][kv]  — B layout for PV   (16 KB)
  __shared__ _Float16 Pl[4][16][64];     // per-wave P bounce                 (8 KB)
  __shared__ float    Ml[2][64][64];     // [buf] mask tile                   (32 KB)

  const int S = 2048;
  const int t = threadIdx.x, lane = t & 31, w = t >> 5;
  const int bh = blockIdx.y;
  const int b = bh >> 5, h = bh & 31, g = h >> 2;   // kv group = head / 4
  const int q0 = blockIdx.x * 64;
  const int ln = lane & 15;
  const int off = (lane & 16) ? 8 : 0;

  const unsigned klbase = (unsigned)(uintptr_t)&Kl[0][0][0];  // buf stride  8192 B
  const unsigned mlbase = (unsigned)(uintptr_t)&Ml[0][0][0];  // buf stride 16384 B
  const size_t   mbase  = (size_t)b * S * S;

  auto stage = [&](int k0, int p) {
    // K tile (64x64 f16): async global->LDS, 512 b128 chunks
#pragma unroll
    for (int i = 0; i < 4; ++i) {
      int c = t + 128 * i;
      int kr = c >> 3, c8 = c & 7;
      async_load_b128(klbase + (unsigned)p * 8192u + kr * 128 + c8 * 16,
                      Kx + (((size_t)b * S + k0 + kr) * 8 + g) * 64 + c8 * 8);
    }
    // mask tile (64x64 f32): async global->LDS, 1024 b128 chunks
#pragma unroll
    for (int i = 0; i < 8; ++i) {
      int c = t + 128 * i;
      int row = c >> 4, c4 = c & 15;
      async_load_b128(mlbase + (unsigned)p * 16384u + row * 256 + c4 * 16,
                      &mask[mbase + (size_t)(q0 + row) * S + k0 + c4 * 4]);
    }
    // V tile (64x64 f16) transposed through VGPRs
    int kr = t >> 1, seg = t & 1;
    const _Float16* vp = Vx + (((size_t)b * S + k0 + kr) * 8 + g) * 64 + seg * 32;
    v8h vv[4];
#pragma unroll
    for (int q = 0; q < 4; ++q) vv[q] = *(const v8h*)(vp + q * 8);
#pragma unroll
    for (int q = 0; q < 4; ++q)
#pragma unroll
      for (int r = 0; r < 8; ++r) Vt[p][seg * 32 + q * 8 + r][kr] = vv[q][r];
  };

  // Q fragments: 16x64 per wave = two 16x32 A-frags, aligned b128 global loads.
  const int qrow = q0 + w * 16 + ln;
  const _Float16* qp = Q + (((size_t)b * S + qrow) * 32 + h) * 64;
  Frag16 qa0, qa1;
  qa0.h[0] = *(const v8h*)(qp + off);
  qa0.h[1] = *(const v8h*)(qp + 16 + off);
  qa1.h[0] = *(const v8h*)(qp + 32 + off);
  qa1.h[1] = *(const v8h*)(qp + 48 + off);

  v8f acc[4] = {};
  float mrow[8], lrow[8];
#pragma unroll
  for (int v = 0; v < 8; ++v) { mrow[v] = -1e30f; lrow[v] = 0.f; }

  stage(0, 0);
  int p = 0;

  for (int k0 = 0; k0 < S; k0 += 64) {
    wait_async0();
    __syncthreads();                     // tile p ready; buffer p^1 free
    if (k0 + 64 < S) stage(k0 + 64, p ^ 1);   // prefetch next KV-block

    // ---- scores: S = Q(16x64) * K^T(64x64) : 4 n-tiles x 2 k-steps ----
    v8f s[4];
#pragma unroll
    for (int nt2 = 0; nt2 < 4; ++nt2) {
      int n = nt2 * 16 + ln;
      Frag16 kb0, kb1;
      kb0.h[0] = *(const v8h*)&Kl[p][n][off];
      kb0.h[1] = *(const v8h*)&Kl[p][n][16 + off];
      kb1.h[0] = *(const v8h*)&Kl[p][n][32 + off];
      kb1.h[1] = *(const v8h*)&Kl[p][n][48 + off];
      v8f z = {};
      z = wmma_f16(qa0.v, kb0.v, z);
      s[nt2] = wmma_f16(qa1.v, kb1.v, z);
    }

    // ---- mask add + online softmax (row = one VGPR across 16 lanes) ----
#pragma unroll
    for (int v = 0; v < 8; ++v) {
      int mr = ((lane & 16) ? 8 : 0) + v;
      float sv[4];
#pragma unroll
      for (int nt2 = 0; nt2 < 4; ++nt2)
        sv[nt2] = s[nt2][v] + Ml[p][w * 16 + mr][nt2 * 16 + ln];
      float sm = fmaxf(fmaxf(sv[0], sv[1]), fmaxf(sv[2], sv[3]));
#pragma unroll
      for (int d = 8; d >= 1; d >>= 1) sm = fmaxf(sm, __shfl_xor(sm, d, 32));
      float mn = fmaxf(mrow[v], sm);
      float pe[4], ps = 0.f;
#pragma unroll
      for (int nt2 = 0; nt2 < 4; ++nt2) { pe[nt2] = __expf(sv[nt2] - mn); ps += pe[nt2]; }
#pragma unroll
      for (int d = 8; d >= 1; d >>= 1) ps += __shfl_xor(ps, d, 32);
      float sc = __expf(mrow[v] - mn);
      lrow[v] = lrow[v] * sc + ps;
      mrow[v] = mn;
#pragma unroll
      for (int dt = 0; dt < 4; ++dt) acc[dt][v] *= sc;
#pragma unroll
      for (int nt2 = 0; nt2 < 4; ++nt2)
        Pl[w][mr][nt2 * 16 + ln] = (_Float16)pe[nt2];
    }
    // Pl is same-wave-only; per-wave LDS ordering makes it visible w/o barrier.

    // ---- PV: acc(16x64) += P(16x64) * V(64x64) : 4 d-tiles x 2 k-steps ----
    Frag16 pa0, pa1;
    pa0.h[0] = *(const v8h*)&Pl[w][ln][off];
    pa0.h[1] = *(const v8h*)&Pl[w][ln][16 + off];
    pa1.h[0] = *(const v8h*)&Pl[w][ln][32 + off];
    pa1.h[1] = *(const v8h*)&Pl[w][ln][48 + off];
#pragma unroll
    for (int dt = 0; dt < 4; ++dt) {
      Frag16 vb0, vb1;
      vb0.h[0] = *(const v8h*)&Vt[p][dt * 16 + ln][off];
      vb0.h[1] = *(const v8h*)&Vt[p][dt * 16 + ln][16 + off];
      vb1.h[0] = *(const v8h*)&Vt[p][dt * 16 + ln][32 + off];
      vb1.h[1] = *(const v8h*)&Vt[p][dt * 16 + ln][48 + off];
      acc[dt] = wmma_f16(pa0.v, vb0.v, acc[dt]);
      acc[dt] = wmma_f16(pa1.v, vb1.v, acc[dt]);
    }
    p ^= 1;
  }

  // ---- normalize and write attention output (f16, head-major columns) ----
#pragma unroll
  for (int v = 0; v < 8; ++v) {
    int mr = ((lane & 16) ? 8 : 0) + v;
    int qr = q0 + w * 16 + mr;
    float inv = 1.f / lrow[v];
    _Float16* op = Out + (((size_t)b * S + qr) * 32 + h) * 64;
#pragma unroll
    for (int dt = 0; dt < 4; ++dt)
      op[dt * 16 + ln] = (_Float16)(acc[dt][v] * inv);
  }
}

// ---------------------------------------------------------------------------
// Launch: Q/K/V projections -> RoPE -> flash attention -> output projection
// ---------------------------------------------------------------------------
extern "C" void kernel_launch(void* const* d_in, const int* in_sizes, int n_in,
                              void* d_out, int out_size, void* d_ws, size_t ws_size,
                              hipStream_t stream) {
  const int B = 2, S = 2048, DIM = 2048;
  const int BS = B * S;                 // 4096
  const int KVD = 512;                  // 8 heads * 64

  const float* q_x     = (const float*)d_in[0];
  const float* kv_x    = (const float*)d_in[1];
  const float* q_freqs = (const float*)d_in[2];
  const float* k_freqs = (const float*)d_in[3];
  const float* mask    = (const float*)d_in[4];
  const float* wq      = (const float*)d_in[5];
  const float* wk      = (const float*)d_in[6];
  const float* wv      = (const float*)d_in[7];
  const float* wo      = (const float*)d_in[8];
  float* out           = (float*)d_out;

  _Float16* Qh = (_Float16*)d_ws;                 // (BS, 2048) f16
  _Float16* Kh = Qh + (size_t)BS * DIM;           // (BS, 512)  f16
  _Float16* Vh = Kh + (size_t)BS * KVD;           // (BS, 512)  f16
  _Float16* Ah = Vh + (size_t)BS * KVD;           // (BS, 2048) f16

  // projections (fp32 in, f16 out, f32 accumulate)
  gemm_tile<false, false><<<dim3(DIM / 64, BS / 128), 256, 0, stream>>>(
      q_x, wq, Qh, BS, DIM, DIM);
  gemm_tile<false, false><<<dim3(KVD / 64, BS / 128), 256, 0, stream>>>(
      kv_x, wk, Kh, BS, KVD, DIM);
  gemm_tile<false, false><<<dim3(KVD / 64, BS / 128), 256, 0, stream>>>(
      kv_x, wv, Vh, BS, KVD, DIM);

  // RoPE (fold 1/sqrt(64) into Q)
  rope_kernel<<<BS, 256, 0, stream>>>(Qh, q_freqs, 32, 0.125f);
  rope_kernel<<<BS, 256, 0, stream>>>(Kh, k_freqs, 8, 1.0f);

  // flash attention: 64 q-rows per block, one (b, head) per y
  attn_kernel<<<dim3(S / 64, B * 32), 128, 0, stream>>>(Qh, Kh, Vh, mask, Ah);

  // output projection (f16 A, fp32 out)
  gemm_tile<true, true><<<dim3(DIM / 64, BS / 128), 256, 0, stream>>>(
      Ah, wo, out, BS, DIM, DIM);
}
